// ContinuousThoughtBlock_18356690223110
// MI455X (gfx1250) — matrix operand order repaired and checked by reading, not compiled, not run on hardware
//
#include <hip/hip_runtime.h>
#include <math.h>

// ---------------------------------------------------------------------------
// Types for CDNA5 WMMA (gfx1250, wave32)
// ---------------------------------------------------------------------------
typedef __attribute__((ext_vector_type(16))) __bf16 bf16x16;
typedef __attribute__((ext_vector_type(4)))  __bf16 bf16x4;
typedef __attribute__((ext_vector_type(8)))  float  f32x8;

#define EPS_LN 1e-6f

// ---------------------------------------------------------------------------
// Elementwise fp32 -> bf16 conversion (x -> A operand). 4 elems per thread.
// ---------------------------------------------------------------------------
__global__ void cvt_bf16_kernel(const float* __restrict__ src,
                                __bf16* __restrict__ dst, long long n) {
  long long i = ((long long)blockIdx.x * blockDim.x + threadIdx.x) * 4;
  if (i + 3 < n) {
    float4 v = *(const float4*)(src + i);
    bf16x4 o;
    o.x = (__bf16)v.x; o.y = (__bf16)v.y; o.z = (__bf16)v.z; o.w = (__bf16)v.w;
    *(bf16x4*)(dst + i) = o;
  }
}

// ---------------------------------------------------------------------------
// gate_w [K,N] fp32 row-major -> Bt [N,K] bf16 row-major (LDS tile transpose)
// block (32,8), grid (N/32, K/32)
// ---------------------------------------------------------------------------
__global__ void transpose_to_bf16_kernel(const float* __restrict__ W,
                                         __bf16* __restrict__ Wt,
                                         int Kdim, int N) {
  __shared__ float tile[32][33];
  int n0 = blockIdx.x * 32;
  int k0 = blockIdx.y * 32;
  int tx = threadIdx.x, ty = threadIdx.y;
  #pragma unroll
  for (int r = 0; r < 32; r += 8)
    tile[ty + r][tx] = W[(size_t)(k0 + ty + r) * N + (n0 + tx)];
  __syncthreads();
  #pragma unroll
  for (int r = 0; r < 32; r += 8)
    Wt[(size_t)(n0 + ty + r) * Kdim + (k0 + tx)] = (__bf16)tile[tx][ty + r];
}

// ---------------------------------------------------------------------------
// Row-wise LayerNorm: one block (256 threads) per row, D <= 2048.
// Row elements cached in registers between reduce and normalize passes.
// rout[d] = (rin[d]-mean)*rsqrt(var+eps)*gamma[d] + beta[d].  In-place safe.
// ---------------------------------------------------------------------------
__global__ void ln_rows_kernel(const float* __restrict__ in,
                               float* __restrict__ out,
                               const float* __restrict__ gamma,
                               const float* __restrict__ beta,
                               int D, long long inStride, long long outStride) {
  __shared__ float red[256];
  long long row = blockIdx.x;
  const float* rin = in + row * inStride;
  float* rout = out + row * outStride;

  float vloc[8];                 // D <= 8 * 256
  int cnt = 0;
  float s = 0.f, ss = 0.f;
  for (int d = threadIdx.x; d < D; d += blockDim.x) {
    float v = rin[d];
    vloc[cnt++] = v;
    s += v; ss += v * v;
  }
  red[threadIdx.x] = s; __syncthreads();
  for (int off = 128; off > 0; off >>= 1) {
    if (threadIdx.x < off) red[threadIdx.x] += red[threadIdx.x + off];
    __syncthreads();
  }
  float mean = red[0] / (float)D;
  __syncthreads();
  red[threadIdx.x] = ss; __syncthreads();
  for (int off = 128; off > 0; off >>= 1) {
    if (threadIdx.x < off) red[threadIdx.x] += red[threadIdx.x + off];
    __syncthreads();
  }
  float var = red[0] / (float)D - mean * mean;
  float inv = rsqrtf(var + EPS_LN);
  cnt = 0;
  for (int d = threadIdx.x; d < D; d += blockDim.x)
    rout[d] = (vloc[cnt++] - mean) * inv * gamma[d] + beta[d];
}

// ---------------------------------------------------------------------------
// Skinny matvec: out[b][c] (+)= act( sum_k in[b][k]*W[k][c] + bias[c] ), b=0..3
// Weight reads fully coalesced (c contiguous); input rows staged in LDS.
// act: 0 = identity, 1 = exact GELU.  accum: 0 = assign, 1 = +=.
// ---------------------------------------------------------------------------
__global__ void matvec4_kernel(const float* __restrict__ in,
                               const float* __restrict__ W,
                               const float* __restrict__ bias,
                               float* __restrict__ out,
                               int Kdim, int N, int act, int accum) {
  __shared__ float s_in[4][1024];
  int c = blockIdx.x * blockDim.x + threadIdx.x;
  float a0 = 0.f, a1 = 0.f, a2 = 0.f, a3 = 0.f;
  for (int k0 = 0; k0 < Kdim; k0 += 1024) {
    for (int i = threadIdx.x; i < 4 * 1024; i += blockDim.x) {
      int b = i >> 10, k = i & 1023;
      s_in[b][k] = in[(size_t)b * Kdim + k0 + k];
    }
    __syncthreads();
    if (c < N) {
      for (int k = 0; k < 1024; ++k) {
        float w = W[(size_t)(k0 + k) * N + c];
        a0 = fmaf(s_in[0][k], w, a0);
        a1 = fmaf(s_in[1][k], w, a1);
        a2 = fmaf(s_in[2][k], w, a2);
        a3 = fmaf(s_in[3][k], w, a3);
      }
    }
    __syncthreads();
  }
  if (c < N) {
    float bb = bias[c];
    float r[4] = {a0 + bb, a1 + bb, a2 + bb, a3 + bb};
    #pragma unroll
    for (int b = 0; b < 4; ++b) {
      float v = r[b];
      if (act == 1) v = 0.5f * v * (1.0f + erff(v * 0.70710678118654752f));
      if (accum) out[(size_t)b * N + c] += v;
      else       out[(size_t)b * N + c]  = v;
    }
  }
}

// ---------------------------------------------------------------------------
// Gate GEMM + fused epilogue (the WMMA kernel):
//   g   = A(bf16)[M,K] @ Bt(bf16)[N,K]^T   (fp32 accum, v_wmma_f32_16x16x32_bf16)
//   out = x + sigmoid(g + gate_b[n]) * bcast[m>>12][n]
// Block: 256 thr = 8 waves, tile 128x128, wave owns 64x32 (4x2 WMMA tiles), BK=32.
// Register double-buffered fragments: loads for step k+32 are issued before the
// WMMAs of step k so the vmem pipe overlaps the matrix pipe.  4-step-ahead
// global_prefetch keeps the L2->L0 stream warm.
// Fragment layouts match the ISA VGPR striping:
//   A 16-bit 16x32: lanes 0-15 K{0..7,16..23}, lanes 16-31 K{8..15,24..31}
//   B 16-bit 32x16: lanes 0-15 K{0..7,16..23}, lanes 16-31 K{8..15,24..31} (Bt row)
// ---------------------------------------------------------------------------
__device__ __forceinline__ void load_frags(const __bf16* __restrict__ Abase,
                                           const __bf16* __restrict__ Bbase,
                                           int K, int kb, int half,
                                           bf16x16 (&afrag)[4],
                                           bf16x16 (&bfrag)[2]) {
  #pragma unroll
  for (int i = 0; i < 4; ++i) {
    const __bf16* p = Abase + (size_t)(16 * i) * K + kb + half * 8;
    union { bf16x16 v; float4 f[2]; } u;
    u.f[0] = *(const float4*)(p);        // K: kb+half*8     .. +7
    u.f[1] = *(const float4*)(p + 16);   // K: kb+16+half*8  .. +7
    afrag[i] = u.v;
  }
  #pragma unroll
  for (int j = 0; j < 2; ++j) {
    const __bf16* p = Bbase + (size_t)(16 * j) * K + kb + half * 8;
    union { bf16x16 v; float4 f[2]; } u;
    u.f[0] = *(const float4*)(p);        // K: kb+half*8     .. +7
    u.f[1] = *(const float4*)(p + 16);   // K: kb+16+half*8  .. +7
    bfrag[j] = u.v;
  }
}

__global__ __launch_bounds__(256)
void gate_gemm_kernel(const __bf16* __restrict__ A,
                      const __bf16* __restrict__ Bt,
                      const float* __restrict__ x,
                      const float* __restrict__ gate_b,
                      const float* __restrict__ bcast,
                      float* __restrict__ out,
                      int M, int N, int K) {
  const int bm = blockIdx.y * 128;
  const int bn = blockIdx.x * 128;

  const int tid  = threadIdx.x;
  const int lane = tid & 31;
  const int wave = tid >> 5;       // 0..7
  const int wm   = wave & 1;       // 2 waves along M
  const int wn   = wave >> 1;      // 4 waves along N
  const int half = lane >> 4;      // 0 or 1 (K-half selector)
  const int l16  = lane & 15;

  const int am0 = bm + wm * 64;    // wave's 64 rows
  const int bn0 = bn + wn * 32;    // wave's 32 cols

  f32x8 acc[4][2] = {};

  const __bf16* Abase = A  + (size_t)(am0 + l16) * K;
  const __bf16* Bbase = Bt + (size_t)(bn0 + l16) * K;

  bf16x16 afrag[2][4];
  bf16x16 bfrag[2][2];
  load_frags(Abase, Bbase, K, 0, half, afrag[0], bfrag[0]);

  int cur = 0;
  for (int kb = 0; kb < K; kb += 32) {
    const int nxt = cur ^ 1;
    if (kb + 32 < K)
      load_frags(Abase, Bbase, K, kb + 32, half, afrag[nxt], bfrag[nxt]);
    if (kb + 128 < K) {            // keep the L2->L0 stream warm (global_prefetch)
      __builtin_prefetch((const void*)(Abase + kb + 128), 0, 1);
      __builtin_prefetch((const void*)(Bbase + kb + 128), 0, 1);
    }
    #pragma unroll
    for (int i = 0; i < 4; ++i)
      #pragma unroll
      for (int j = 0; j < 2; ++j)
        acc[i][j] = __builtin_amdgcn_wmma_f32_16x16x32_bf16(
            false, afrag[cur][i], false, bfrag[cur][j],
            (short)0, acc[i][j], false, false);
    cur = nxt;
  }

  // Fused epilogue: C-tile layout -> (m, n); out = x + sigmoid(g+gb)*bcast
  #pragma unroll
  for (int i = 0; i < 4; ++i) {
    const int mrow = am0 + 16 * i + half * 8;       // + e
    #pragma unroll
    for (int j = 0; j < 2; ++j) {
      const int n  = bn0 + 16 * j + l16;
      const float gb = gate_b[n];
      const size_t base = (size_t)mrow * N + n;
      #pragma unroll
      for (int e = 0; e < 8; ++e) {
        const float g = acc[i][j][e] + gb;
        const float sg = 1.0f / (1.0f + expf(-g));
        const int b = (mrow + e) >> 12;             // L = 4096
        const size_t idx = base + (size_t)e * N;
        out[idx] = x[idx] + sg * bcast[(size_t)b * N + n];
      }
    }
  }
}

// ---------------------------------------------------------------------------
// Host launcher
// ---------------------------------------------------------------------------
extern "C" void kernel_launch(void* const* d_in, const int* in_sizes, int n_in,
                              void* d_out, int out_size, void* d_ws, size_t ws_size,
                              hipStream_t stream) {
  (void)in_sizes; (void)n_in; (void)out_size; (void)ws_size;
  const int B = 4, L = 4096, D = 2048, H = 8192;
  const long long M = (long long)B * L;   // 16384

  const float* x        = (const float*)d_in[0];
  const float* in_gamma = (const float*)d_in[1];
  const float* in_beta  = (const float*)d_in[2];
  const float* agg_w    = (const float*)d_in[3];
  const float* agg_b    = (const float*)d_in[4];
  const float* pn_gamma = (const float*)d_in[5];
  const float* pn_beta  = (const float*)d_in[6];
  const float* d1_w     = (const float*)d_in[7];
  const float* d1_b     = (const float*)d_in[8];
  const float* d2_w     = (const float*)d_in[9];
  const float* d2_b     = (const float*)d_in[10];
  const float* bc_w     = (const float*)d_in[11];
  const float* bc_b     = (const float*)d_in[12];
  const float* gate_w   = (const float*)d_in[13];
  const float* gate_b   = (const float*)d_in[14];
  const float* out_gamma= (const float*)d_in[15];
  const float* out_beta = (const float*)d_in[16];
  float* out = (float*)d_out;

  // Workspace carve-out (256B aligned): ~76 MB total
  char* ws = (char*)d_ws;
  size_t off = 0;
  auto take = [&](size_t bytes) -> char* {
    off = (off + 255) & ~(size_t)255;
    char* p = ws + off;
    off += bytes;
    return p;
  };
  __bf16* A16     = (__bf16*)take((size_t)M * D * 2);       // 64 MB
  __bf16* Bt16    = (__bf16*)take((size_t)D * D * 2);       //  8 MB
  float*  ln_last = (float*)take((size_t)4 * D * 4);
  float*  thought = (float*)take((size_t)4 * D * 4);
  float*  tln     = (float*)take((size_t)4 * D * 4);
  float*  hbuf    = (float*)take((size_t)4 * H * 4);
  float*  bcast   = (float*)take((size_t)4 * D * 4);

  // 1) Convert x -> bf16 A operand
  {
    long long n = M * D;                      // 33,554,432 (mult of 4)
    int blocks = (int)((n / 4 + 255) / 256);
    cvt_bf16_kernel<<<blocks, 256, 0, stream>>>(x, A16, n);
  }
  // 2) Transpose+convert gate_w -> Bt (bf16, [N,K])
  {
    dim3 tb(32, 8), tg(D / 32, D / 32);
    transpose_to_bf16_kernel<<<tg, tb, 0, stream>>>(gate_w, Bt16, D, D);
  }
  // 3) LN of last-position rows only (x_norm[:, -1, :])
  ln_rows_kernel<<<4, 256, 0, stream>>>(x + (size_t)(L - 1) * D, ln_last,
                                        in_gamma, in_beta, D,
                                        (long long)L * D, (long long)D);
  // 4) thought = ln_last @ agg_w + agg_b
  matvec4_kernel<<<D / 256, 256, 0, stream>>>(ln_last, agg_w, agg_b, thought,
                                              D, D, /*act*/0, /*accum*/0);
  // 5) 4 refinement steps (memory-bound matvecs; weights L2-resident)
  for (int it = 0; it < 4; ++it) {
    ln_rows_kernel<<<4, 256, 0, stream>>>(thought, tln, pn_gamma, pn_beta, D,
                                          (long long)D, (long long)D);
    matvec4_kernel<<<H / 256, 256, 0, stream>>>(tln, d1_w, d1_b, hbuf,
                                                D, H, /*gelu*/1, /*accum*/0);
    matvec4_kernel<<<D / 256, 256, 0, stream>>>(hbuf, d2_w, d2_b, thought,
                                                H, D, /*act*/0, /*accum*/1);
  }
  // 6) broadcast = thought @ bc_w + bc_b
  matvec4_kernel<<<D / 256, 256, 0, stream>>>(thought, bc_w, bc_b, bcast,
                                              D, D, /*act*/0, /*accum*/0);
  // 7) WMMA gate GEMM + fused sigmoid/residual epilogue -> out (pre-LN)
  {
    dim3 grid(D / 128, (unsigned)(M / 128));  // (16, 128)
    gate_gemm_kernel<<<grid, 256, 0, stream>>>(A16, Bt16, x, gate_b, bcast,
                                               out, (int)M, D, D);
  }
  // 8) Final LN over rows, in place
  ln_rows_kernel<<<(unsigned)M, 256, 0, stream>>>(out, out, out_gamma, out_beta,
                                                  D, (long long)D, (long long)D);
}
